// BaseKernel_46617575030959
// MI455X (gfx1250) — matrix operand order repaired and checked by reading, not compiled, not run on hardware
//
#include <hip/hip_runtime.h>
#include <hip/hip_bf16.h>
#include <math.h>

#define NATOMS 16384
#define NMOL   1024
#define REP    512
#define PROJ   256
#define NFEAT  4096
#define NELEM  4
#define OUTC   (NFEAT + 1)            // 4097 output columns
#define SCALE  0.022097086912079608f  // sqrt(2/NFEAT)
#define LLAMBDA 1e-6f

typedef __attribute__((ext_vector_type(16))) __bf16 bf16x16;
typedef __attribute__((ext_vector_type(8)))  __bf16 bf16x8;
typedef __attribute__((ext_vector_type(8)))  float  f32x8;

// ---------------- helpers ----------------

__device__ __forceinline__ unsigned short f32_to_bf16(float f) {
  union { float f; unsigned u; } v; v.f = f;
  unsigned u = v.u;
  u += 0x7FFFu + ((u >> 16) & 1u);   // round-to-nearest-even
  return (unsigned short)(u >> 16);
}

// A-operand (16-bit, 16x32 MxK): lane l holds row M=l%16.
// lanes 0-15: K = {k0+0..7} U {k0+16..23}; lanes 16-31: K = {k0+8..15} U {k0+24..31}
__device__ __forceinline__ bf16x16 load_a16(const unsigned short* rowPtr, int k0, int lane) {
  const int off = (lane & 16) ? 8 : 0;
  const bf16x8 lo = *(const bf16x8*)(rowPtr + k0 + off);
  const bf16x8 hi = *(const bf16x8*)(rowPtr + k0 + off + 16);
  bf16x16 r;
#pragma unroll
  for (int i = 0; i < 8; ++i) { r[i] = lo[i]; r[i + 8] = hi[i]; }
  return r;
}

// B-operand (16-bit, 32x16 KxN): lane l holds column N=l%16:
// lanes 0-15: K = {k0+0..7} U {k0+16..23}; lanes 16-31: K = {k0+8..15} U {k0+24..31}
// colPtr points at a K-contiguous row of the transposed matrix.
__device__ __forceinline__ bf16x16 load_b16(const unsigned short* colPtr, int k0, int lane) {
  const int off = (lane & 16) ? 8 : 0;
  const bf16x8 lo = *(const bf16x8*)(colPtr + k0 + off);
  const bf16x8 hi = *(const bf16x8*)(colPtr + k0 + off + 16);
  bf16x16 r;
#pragma unroll
  for (int i = 0; i < 8; ++i) { r[i] = lo[i]; r[i + 8] = hi[i]; }
  return r;
}

__device__ __forceinline__ f32x8 wmma_bf16(bf16x16 a, bf16x16 b, f32x8 c) {
  return __builtin_amdgcn_wmma_f32_16x16x32_bf16(false, a, false, b, (short)0, c, false, false);
}

// ---------------- kernel 1: convert / transpose / zero ----------------

__global__ void k_prep(const float* __restrict__ gto,
                       const float* __restrict__ reductors,
                       const float* __restrict__ W,
                       unsigned short* __restrict__ gtoB,     // [NATOMS][REP]
                       unsigned short* __restrict__ redT,     // [NELEM][PROJ][REP]
                       unsigned short* __restrict__ WT,       // [NELEM][NFEAT][PROJ]
                       float* __restrict__ Ztrain) {          // [NMOL][NFEAT]
  const int stride = gridDim.x * blockDim.x;
  for (int i = blockIdx.x * blockDim.x + threadIdx.x; i < NATOMS * REP; i += stride)
    gtoB[i] = f32_to_bf16(gto[i]);
  for (int i = blockIdx.x * blockDim.x + threadIdx.x; i < NELEM * PROJ * REP; i += stride) {
    int e = i / (PROJ * REP), rem = i % (PROJ * REP);
    int p = rem / REP, k = rem % REP;
    redT[i] = f32_to_bf16(reductors[(e * REP + k) * PROJ + p]);
  }
  for (int i = blockIdx.x * blockDim.x + threadIdx.x; i < NELEM * NFEAT * PROJ; i += stride) {
    int e = i / (NFEAT * PROJ), rem = i % (NFEAT * PROJ);
    int n = rem / PROJ, k = rem % PROJ;
    WT[i] = f32_to_bf16(W[(e * PROJ + k) * NFEAT + n]);
  }
  for (int i = blockIdx.x * blockDim.x + threadIdx.x; i < NMOL * NFEAT; i += stride)
    Ztrain[i] = 0.0f;
}

// ---------------- kernel 2: deterministic element bucketing ----------------

__global__ void k_bucket(const int* __restrict__ charges,
                         int* __restrict__ list,    // [NELEM][NATOMS]
                         int* __restrict__ cnt) {   // [NELEM]
  const int a = blockIdx.x * blockDim.x + threadIdx.x;
  if (a < NATOMS) {
    const int c = charges[a];
    int pos = 0;
    for (int i = 0; i < a; ++i) pos += (charges[i] == c) ? 1 : 0;
    list[c * NATOMS + pos] = a;
  }
  if (a < NELEM) {
    int n = 0;
    for (int i = 0; i < NATOMS; ++i) n += (charges[i] == a) ? 1 : 0;
    cnt[a] = n;
  }
}

// ---------------- kernel 3: fused projection + random features + scatter ----------------
// One block (8 waves) per 16-atom tile of a single element expert.

__global__ void __launch_bounds__(256)
k_fused(const unsigned short* __restrict__ gtoB,
        const unsigned short* __restrict__ redT,
        const unsigned short* __restrict__ WT,
        const float* __restrict__ bvec,            // [NELEM][NFEAT]
        const int* __restrict__ molIDs,
        const int* __restrict__ list,
        const int* __restrict__ cnt,
        float* __restrict__ Ztrain) {
  const int e = blockIdx.x >> 10;
  const int t = blockIdx.x & 1023;
  const int ne = cnt[e];
  if (t * 16 >= ne) return;

  __shared__ unsigned short sProj[16 * 264];   // 16x256 bf16, padded stride (528B rows)
  __shared__ int sAtom[16];
  __shared__ int sMol[16];

  const int tid = threadIdx.x;
  if (tid < 16) {
    const int i = t * 16 + tid;
    const int a = (i < ne) ? list[e * NATOMS + i] : -1;
    sAtom[tid] = a;
    sMol[tid] = (a >= 0) ? molIDs[a] : 0;
  }
  __syncthreads();

  const int wave = tid >> 5;
  const int lane = tid & 31;
  const int nlo  = lane & 15;            // N (or M-row select) within tile
  const int mhi  = (lane >> 4) << 3;     // C-layout: rows r and r+8

  // ---- phase 1: proj tile [16 x 256] = gto_rows[16 x 512] @ red[e] ----
  {
    const int aIdx = sAtom[nlo];
    const unsigned short* aRow = gtoB + (size_t)(aIdx < 0 ? 0 : aIdx) * REP;
#pragma unroll
    for (int half = 0; half < 2; ++half) {
      const int p0 = wave * 16 + half * 128;
      const unsigned short* bRow = redT + ((size_t)(e * PROJ + p0 + nlo)) * REP;
      f32x8 acc = {0.f, 0.f, 0.f, 0.f, 0.f, 0.f, 0.f, 0.f};
      for (int k0 = 0; k0 < REP; k0 += 32) {
        bf16x16 av = load_a16(aRow, k0, lane);
        bf16x16 bv = load_b16(bRow, k0, lane);
        acc = wmma_bf16(av, bv, acc);
      }
      const int n = p0 + nlo;
#pragma unroll
      for (int r = 0; r < 8; ++r)
        sProj[(r + mhi) * 264 + n] = f32_to_bf16(acc[r]);
    }
  }
  __syncthreads();

  // ---- phase 2: feats [16 x 4096] = projTile @ W[e], cos, scatter-add ----
  const unsigned short* aRow = sProj + (size_t)nlo * 264;
  for (int nt = wave; nt < NFEAT / 16; nt += 8) {
    const int n0 = nt * 16;
    const unsigned short* bRow = WT + ((size_t)(e * NFEAT + n0 + nlo)) * PROJ;
    __builtin_prefetch((const void*)(bRow + PROJ * 16 * 8), 0, 1);  // next N-tile for this wave
    f32x8 acc = {0.f, 0.f, 0.f, 0.f, 0.f, 0.f, 0.f, 0.f};
#pragma unroll
    for (int k0 = 0; k0 < PROJ; k0 += 32) {
      bf16x16 av = load_a16(aRow, k0, lane);
      bf16x16 bv = load_b16(bRow, k0, lane);
      acc = wmma_bf16(av, bv, acc);
    }
    const float bias = bvec[e * NFEAT + n0 + nlo];
#pragma unroll
    for (int r = 0; r < 8; ++r) {
      const int M = r + mhi;
      if (sAtom[M] >= 0) {
        const float fv = SCALE * cosf(acc[r] + bias);
        unsafeAtomicAdd(&Ztrain[(size_t)sMol[M] * NFEAT + n0 + nlo], fv);
      }
    }
  }
}

// ---------------- kernel 4: Ztrain -> bf16 transpose [NFEAT][NMOL] ----------------

__global__ void k_transZ(const float* __restrict__ Ztrain,
                         unsigned short* __restrict__ ZtrainT) {
  const int stride = gridDim.x * blockDim.x;
  for (int i = blockIdx.x * blockDim.x + threadIdx.x; i < NFEAT * NMOL; i += stride) {
    const int f = i >> 10, m = i & (NMOL - 1);
    ZtrainT[i] = f32_to_bf16(Ztrain[(size_t)m * NFEAT + f]);
  }
}

// ---------------- kernel 5: ZtY column (full f32 precision) ----------------

__global__ void k_zty(const float* __restrict__ Ztrain,
                      const float* __restrict__ Y,
                      float* __restrict__ out) {
  const int f = blockIdx.x * blockDim.x + threadIdx.x;
  if (f >= NFEAT) return;
  float s = 0.f;
  for (int m = 0; m < NMOL; ++m)
    s += Ztrain[(size_t)m * NFEAT + f] * Y[m];
  out[(size_t)f * OUTC + NFEAT] = s;
}

// ---------------- kernel 6: ZTZ = Z^T Z + lambda*I via WMMA ----------------
// ZtrainT[f][m] serves as BOTH operands: A row i and B column j are K-contiguous.

__global__ void __launch_bounds__(256)
k_ztz(const unsigned short* __restrict__ ZtrainT,
      float* __restrict__ out) {
  const int tid  = threadIdx.x;
  const int lane = tid & 31;
  const int wg   = blockIdx.x * 8 + (tid >> 5);   // 65536 wave-tiles
  const int it = wg >> 8, jt = wg & 255;
  const int i0 = it * 16, j0 = jt * 16;
  const int nlo = lane & 15;
  const int mhi = (lane >> 4) << 3;

  const unsigned short* aRow = ZtrainT + (size_t)(i0 + nlo) * NMOL;
  const unsigned short* bRow = ZtrainT + (size_t)(j0 + nlo) * NMOL;

  f32x8 acc = {0.f, 0.f, 0.f, 0.f, 0.f, 0.f, 0.f, 0.f};
#pragma unroll 4
  for (int k0 = 0; k0 < NMOL; k0 += 32) {
    bf16x16 av = load_a16(aRow, k0, lane);
    bf16x16 bv = load_b16(bRow, k0, lane);
    acc = wmma_bf16(av, bv, acc);
  }
#pragma unroll
  for (int r = 0; r < 8; ++r) {
    const int row = i0 + r + mhi;
    const int col = j0 + nlo;
    float v = acc[r];
    if (row == col) v += LLAMBDA;
    out[(size_t)row * OUTC + col] = v;
  }
}

// ---------------- launch ----------------

extern "C" void kernel_launch(void* const* d_in, const int* in_sizes, int n_in,
                              void* d_out, int out_size, void* d_ws, size_t ws_size,
                              hipStream_t stream) {
  const float* gto       = (const float*)d_in[0];
  const float* reductors = (const float*)d_in[1];
  const float* W         = (const float*)d_in[2];
  const float* bvec      = (const float*)d_in[3];
  const float* Y         = (const float*)d_in[4];
  const int*   charges   = (const int*)d_in[5];
  const int*   molIDs    = (const int*)d_in[6];
  float* out = (float*)d_out;

  char* ws = (char*)d_ws;
  size_t off = 0;
  unsigned short* gtoB    = (unsigned short*)(ws + off); off += (size_t)NATOMS * REP * 2;        // 16.78 MB
  unsigned short* redT    = (unsigned short*)(ws + off); off += (size_t)NELEM * PROJ * REP * 2;  //  1.05 MB
  unsigned short* WT      = (unsigned short*)(ws + off); off += (size_t)NELEM * NFEAT * PROJ * 2;//  8.39 MB
  float*          Ztrain  = (float*)(ws + off);          off += (size_t)NMOL * NFEAT * 4;        // 16.78 MB
  unsigned short* ZtrainT = (unsigned short*)(ws + off); off += (size_t)NFEAT * NMOL * 2;        //  8.39 MB
  int*            list    = (int*)(ws + off);            off += (size_t)NELEM * NATOMS * 4;      //  0.26 MB
  int*            cnt     = (int*)(ws + off);            off += 64;

  k_prep  <<<8192, 256, 0, stream>>>(gto, reductors, W, gtoB, redT, WT, Ztrain);
  k_bucket<<<NATOMS / 256, 256, 0, stream>>>(charges, list, cnt);
  k_fused <<<NELEM * 1024, 256, 0, stream>>>(gtoB, redT, WT, bvec, molIDs, list, cnt, Ztrain);
  k_transZ<<<4096, 256, 0, stream>>>(Ztrain, ZtrainT);
  k_zty   <<<NFEAT / 256, 256, 0, stream>>>(Ztrain, Y, out);
  k_ztz   <<<(256 * 256) / 8, 256, 0, stream>>>(ZtrainT, out);
}
